// SimpleGNN_59098749993602
// MI455X (gfx1250) — matrix operand order repaired and checked by reading, not compiled, not run on hardware
//
#include <hip/hip_runtime.h>
#include <hip/hip_bf16.h>
#include <math.h>

#define NN 50000
#define NE 800000
#define HID 64
#define TWO_PI 6.283185307179586f

typedef _Float16 v16h __attribute__((ext_vector_type(16)));
typedef _Float16 v8h  __attribute__((ext_vector_type(8)));
typedef float    v8f  __attribute__((ext_vector_type(8)));

__device__ __forceinline__ v16h cat8(v8h lo, v8h hi) {
  return __builtin_shufflevector(lo, hi, 0,1,2,3,4,5,6,7,8,9,10,11,12,13,14,15);
}

__device__ __forceinline__ v8f wmma16(v16h a, v16h b, v8f c) {
  return __builtin_amdgcn_wmma_f32_16x16x32_f16(false, a, false, b, (short)0, c, false, false);
}

// packed f16 atomic add to global (CDNA5 GLOBAL_ATOMIC_PK_ADD_F16, no return)
__device__ __forceinline__ void atomic_pk_add_f16(_Float16* addr, float lo, float hi) {
  union { _Float16 h[2]; unsigned int u; } cv;
  cv.h[0] = (_Float16)lo;
  cv.h[1] = (_Float16)hi;
  asm volatile("global_atomic_pk_add_f16 %0, %1, off"
               :: "v"((unsigned long long)(uintptr_t)addr), "v"(cv.u)
               : "memory");
}

__device__ __forceinline__ void block_copy_h(const _Float16* __restrict__ g, _Float16* s, int n) {
  for (int i = threadIdx.x * 8; i < n; i += blockDim.x * 8)
    *(v8h*)(s + i) = *(const v8h*)(g + i);
}
__device__ __forceinline__ void block_copy_f(const float* __restrict__ g, float* s, int n) {
  for (int i = threadIdx.x; i < n; i += blockDim.x) s[i] = g[i];
}

// ---------------------------------------------------------------------------
// Convert row-major f32 weight [Ktiles*32 x 64] into WMMA B-fragment f16 layout:
// frag[((kt*4+nt)*32 + lane)*16 + i] = W[kt*32 + (lane<16?0:8) + (i<8?i:i+8)][nt*16 + lane%16]
// ---------------------------------------------------------------------------
__global__ void convert_bfrag(const float* __restrict__ W, _Float16* __restrict__ frag,
                              int Ktiles) {
  int tid = blockIdx.x * blockDim.x + threadIdx.x;
  int total = Ktiles * 4 * 32 * 16;
  if (tid >= total) return;
  int i    = tid & 15;
  int lane = (tid >> 4) & 31;
  int nt   = (tid >> 9) & 3;
  int kt   = tid >> 11;
  int k = kt * 32 + ((lane >> 4) ? 8 : 0) + ((i < 8) ? i : (i + 8));
  int n = nt * 16 + (lane & 15);
  frag[tid] = (_Float16)W[k * HID + n];
}

__global__ void zero_f32(float* __restrict__ p, int n) {
  int t = blockIdx.x * blockDim.x + threadIdx.x;
  if (t < n) p[t] = 0.f;
}

__global__ void count_edges(const long long* __restrict__ dst, float* __restrict__ cnt) {
  int t = blockIdx.x * blockDim.x + threadIdx.x;
  if (t < NE) unsafeAtomicAdd(&cnt[(int)dst[t]], 1.f);
}

__global__ void finalize_inv(float* __restrict__ c, int n) {
  int t = blockIdx.x * blockDim.x + threadIdx.x;
  if (t < n) c[t] = 1.f / fmaxf(c[t], 1.f);
}

// ---------------------------------------------------------------------------
// Node encoder: 5 -> 64 -> 64, relu both, f32 VALU (tiny), writes h as f16
// ---------------------------------------------------------------------------
__global__ __launch_bounds__(256) void enc_kernel(
    const float* __restrict__ x,
    const float* __restrict__ W1, const float* __restrict__ b1,
    const float* __restrict__ W2, const float* __restrict__ b2,
    _Float16* __restrict__ hout) {
  __shared__ float sW1[5 * HID], sB1[HID], sW2[HID * HID], sB2[HID];
  block_copy_f(W1, sW1, 5 * HID);
  block_copy_f(b1, sB1, HID);
  block_copy_f(W2, sW2, HID * HID);
  block_copy_f(b2, sB2, HID);
  __syncthreads();
  int node = blockIdx.x * blockDim.x + threadIdx.x;
  if (node >= NN) return;
  float xv[5];
#pragma unroll
  for (int k = 0; k < 5; ++k) xv[k] = x[node * 5 + k];
  float h1[HID];
#pragma unroll
  for (int j = 0; j < HID; ++j) {
    float v = sB1[j];
#pragma unroll
    for (int k = 0; k < 5; ++k) v = fmaf(xv[k], sW1[k * HID + j], v);
    h1[j] = fmaxf(v, 0.f);
  }
  for (int j = 0; j < HID; ++j) {
    float v = sB2[j];
#pragma unroll
    for (int k = 0; k < HID; ++k) v = fmaf(h1[k], sW2[k * HID + j], v);
    hout[(size_t)node * HID + j] = (_Float16)fmaxf(v, 0.f);
  }
}

// ---------------------------------------------------------------------------
// Message layer: 1 wave = 16 edges. A = [h[src] | relu(ef*eW+eb)] (16x128 f16)
// hidden = relu(A @ W1 + b1) staged via per-wave LDS tile; msg = relu(h' @ W2 + b2)
// scatter-add into f16 agg[dst] with packed-f16 atomics (2 cols per op).
// ---------------------------------------------------------------------------
__global__ __launch_bounds__(256) void msg_kernel(
    const _Float16* __restrict__ h,
    const long long* __restrict__ src, const long long* __restrict__ dst,
    const float* __restrict__ efeat,
    const float* __restrict__ eW, const float* __restrict__ eb,
    const _Float16* __restrict__ W1f, const float* __restrict__ b1,
    const _Float16* __restrict__ W2f, const float* __restrict__ b2,
    _Float16* __restrict__ agg) {
  __shared__ __align__(16) _Float16 sW1[4 * 4 * 32 * 16];
  __shared__ __align__(16) _Float16 sW2[2 * 4 * 32 * 16];
  __shared__ __align__(16) _Float16 sAct[8 * 16 * HID];
  __shared__ float sEW[HID], sEB[HID];

  block_copy_h(W1f, sW1, 4 * 4 * 32 * 16);
  block_copy_h(W2f, sW2, 2 * 4 * 32 * 16);
  block_copy_f(eW, sEW, HID);
  block_copy_f(eb, sEB, HID);
  __syncthreads();

  const int lane = threadIdx.x & 31;
  const int wave = threadIdx.x >> 5;
  const int lane16 = lane & 15;
  const int hi = lane >> 4;
  const int kbA = hi * 8;
  const int tile = blockIdx.x * 8 + wave;  // exact: NE/16 tiles, grid = NE/128

  const int erow = tile * 16 + lane16;
  const long long s = src[erow];
  const _Float16* hp = h + (size_t)s * HID;
  v16h a0 = cat8(*(const v8h*)(hp + kbA),      *(const v8h*)(hp + kbA + 16));
  v16h a1 = cat8(*(const v8h*)(hp + 32 + kbA), *(const v8h*)(hp + 32 + kbA + 16));

  const float ef = efeat[erow];
  v16h a2, a3;
#pragma unroll
  for (int i = 0; i < 16; ++i) {
    int j = kbA + ((i < 8) ? i : (8 + i));
    a2[i] = (_Float16)fmaxf(fmaf(ef, sEW[j], sEB[j]), 0.f);
    a3[i] = (_Float16)fmaxf(fmaf(ef, sEW[32 + j], sEB[32 + j]), 0.f);
  }

  _Float16* act = sAct + wave * (16 * HID);
  v16h afr[4] = {a0, a1, a2, a3};
#pragma unroll
  for (int nt = 0; nt < 4; ++nt) {
    float bv = b1[nt * 16 + lane16];
    v8f acc = {bv, bv, bv, bv, bv, bv, bv, bv};
#pragma unroll
    for (int kt = 0; kt < 4; ++kt) {
      const _Float16* w = sW1 + ((kt * 4 + nt) * 32 + lane) * 16;
      acc = wmma16(afr[kt], cat8(*(const v8h*)w, *(const v8h*)(w + 8)), acc);
    }
#pragma unroll
    for (int r = 0; r < 8; ++r) {
      int m = hi ? (r + 8) : r;
      act[m * HID + nt * 16 + lane16] = (_Float16)fmaxf(acc[r], 0.f);
    }
  }
  __syncthreads();

  const _Float16* arow = act + lane16 * HID;
  v16h cfr[2];
  cfr[0] = cat8(*(const v8h*)(arow + kbA),      *(const v8h*)(arow + kbA + 16));
  cfr[1] = cat8(*(const v8h*)(arow + 32 + kbA), *(const v8h*)(arow + 32 + kbA + 16));

  int dn[8];
#pragma unroll
  for (int r = 0; r < 8; ++r) {
    int m = hi ? (r + 8) : r;
    dn[r] = (int)dst[tile * 16 + m];
  }

#pragma unroll
  for (int nt = 0; nt < 4; ++nt) {
    float bv = b2[nt * 16 + lane16];
    v8f acc = {bv, bv, bv, bv, bv, bv, bv, bv};
#pragma unroll
    for (int kt = 0; kt < 2; ++kt) {
      const _Float16* w = sW2 + ((kt * 4 + nt) * 32 + lane) * 16;
      acc = wmma16(cfr[kt], cat8(*(const v8h*)w, *(const v8h*)(w + 8)), acc);
    }
    // Within a given acc element r, lanes 0-15 (and 16-31) all hold the SAME
    // row at col = lane16, so lane pairs hold adjacent columns: pack two
    // halves and let even lanes issue one packed-f16 atomic.
#pragma unroll
    for (int r = 0; r < 8; ++r) {
      float mine = fmaxf(acc[r], 0.f);
      float nb = __shfl_xor(mine, 1, 32);
      if (!(lane & 1)) {
        atomic_pk_add_f16(agg + (size_t)dn[r] * HID + nt * 16 + lane16, mine, nb);
      }
    }
  }
}

// ---------------------------------------------------------------------------
// Update layer: 1 wave = 16 nodes. A = [h | agg_f16 * inv_denom], writes h_next
// ---------------------------------------------------------------------------
__global__ __launch_bounds__(256) void upd_kernel(
    const _Float16* __restrict__ h,
    const _Float16* __restrict__ agg, const float* __restrict__ invd,
    const _Float16* __restrict__ W1f, const float* __restrict__ b1,
    const _Float16* __restrict__ W2f, const float* __restrict__ b2,
    _Float16* __restrict__ hout) {
  __shared__ __align__(16) _Float16 sW1[4 * 4 * 32 * 16];
  __shared__ __align__(16) _Float16 sW2[2 * 4 * 32 * 16];
  __shared__ __align__(16) _Float16 sAct[8 * 16 * HID];

  block_copy_h(W1f, sW1, 4 * 4 * 32 * 16);
  block_copy_h(W2f, sW2, 2 * 4 * 32 * 16);
  __syncthreads();

  const int lane = threadIdx.x & 31;
  const int wave = threadIdx.x >> 5;
  const int lane16 = lane & 15;
  const int hi = lane >> 4;
  const int kbA = hi * 8;
  int tile = blockIdx.x * 8 + wave;
  const bool active = tile < (NN / 16);
  if (!active) tile = 0;

  const int node = tile * 16 + lane16;
  const _Float16* hp = h + (size_t)node * HID;
  v16h a0 = cat8(*(const v8h*)(hp + kbA),      *(const v8h*)(hp + kbA + 16));
  v16h a1 = cat8(*(const v8h*)(hp + 32 + kbA), *(const v8h*)(hp + 32 + kbA + 16));

  const float inv = invd[node];
  const _Float16* ap = agg + (size_t)node * HID;
  v16h t2 = cat8(*(const v8h*)(ap + kbA),      *(const v8h*)(ap + kbA + 16));
  v16h t3 = cat8(*(const v8h*)(ap + 32 + kbA), *(const v8h*)(ap + 32 + kbA + 16));
  v16h a2, a3;
#pragma unroll
  for (int i = 0; i < 16; ++i) {
    a2[i] = (_Float16)((float)t2[i] * inv);
    a3[i] = (_Float16)((float)t3[i] * inv);
  }

  _Float16* act = sAct + wave * (16 * HID);
  v16h afr[4] = {a0, a1, a2, a3};
#pragma unroll
  for (int nt = 0; nt < 4; ++nt) {
    float bv = b1[nt * 16 + lane16];
    v8f acc = {bv, bv, bv, bv, bv, bv, bv, bv};
#pragma unroll
    for (int kt = 0; kt < 4; ++kt) {
      const _Float16* w = sW1 + ((kt * 4 + nt) * 32 + lane) * 16;
      acc = wmma16(afr[kt], cat8(*(const v8h*)w, *(const v8h*)(w + 8)), acc);
    }
#pragma unroll
    for (int r = 0; r < 8; ++r) {
      int m = hi ? (r + 8) : r;
      act[m * HID + nt * 16 + lane16] = (_Float16)fmaxf(acc[r], 0.f);
    }
  }
  __syncthreads();

  const _Float16* arow = act + lane16 * HID;
  v16h cfr[2];
  cfr[0] = cat8(*(const v8h*)(arow + kbA),      *(const v8h*)(arow + kbA + 16));
  cfr[1] = cat8(*(const v8h*)(arow + 32 + kbA), *(const v8h*)(arow + 32 + kbA + 16));

#pragma unroll
  for (int nt = 0; nt < 4; ++nt) {
    float bv = b2[nt * 16 + lane16];
    v8f acc = {bv, bv, bv, bv, bv, bv, bv, bv};
#pragma unroll
    for (int kt = 0; kt < 2; ++kt) {
      const _Float16* w = sW2 + ((kt * 4 + nt) * 32 + lane) * 16;
      acc = wmma16(cfr[kt], cat8(*(const v8h*)w, *(const v8h*)(w + 8)), acc);
    }
    if (active) {
#pragma unroll
      for (int r = 0; r < 8; ++r) {
        int m = hi ? (r + 8) : r;
        hout[(size_t)(tile * 16 + m) * HID + nt * 16 + lane16] = (_Float16)fmaxf(acc[r], 0.f);
      }
    }
  }
}

// ---------------------------------------------------------------------------
// Output head: hidden = relu(h @ outW1 + b1) via WMMA, then 64->3 VALU + sigmoid
// ---------------------------------------------------------------------------
__global__ __launch_bounds__(256) void head_kernel(
    const _Float16* __restrict__ h,
    const _Float16* __restrict__ W1f, const float* __restrict__ b1,
    const float* __restrict__ W2, const float* __restrict__ b2,
    float* __restrict__ out) {
  __shared__ __align__(16) _Float16 sW1[2 * 4 * 32 * 16];
  __shared__ __align__(16) _Float16 sAct[8 * 16 * HID];
  __shared__ float sW2[HID * 3], sB2[3];

  block_copy_h(W1f, sW1, 2 * 4 * 32 * 16);
  block_copy_f(W2, sW2, HID * 3);
  block_copy_f(b2, sB2, 3);
  __syncthreads();

  const int lane = threadIdx.x & 31;
  const int wave = threadIdx.x >> 5;
  const int lane16 = lane & 15;
  const int hi = lane >> 4;
  const int kbA = hi * 8;
  int tile = blockIdx.x * 8 + wave;
  const bool active = tile < (NN / 16);
  if (!active) tile = 0;

  const _Float16* hp = h + (size_t)(tile * 16 + lane16) * HID;
  v16h afr[2];
  afr[0] = cat8(*(const v8h*)(hp + kbA),      *(const v8h*)(hp + kbA + 16));
  afr[1] = cat8(*(const v8h*)(hp + 32 + kbA), *(const v8h*)(hp + 32 + kbA + 16));

  _Float16* act = sAct + wave * (16 * HID);
#pragma unroll
  for (int nt = 0; nt < 4; ++nt) {
    float bv = b1[nt * 16 + lane16];
    v8f acc = {bv, bv, bv, bv, bv, bv, bv, bv};
#pragma unroll
    for (int kt = 0; kt < 2; ++kt) {
      const _Float16* w = sW1 + ((kt * 4 + nt) * 32 + lane) * 16;
      acc = wmma16(afr[kt], cat8(*(const v8h*)w, *(const v8h*)(w + 8)), acc);
    }
#pragma unroll
    for (int r = 0; r < 8; ++r) {
      int m = hi ? (r + 8) : r;
      act[m * HID + nt * 16 + lane16] = (_Float16)fmaxf(acc[r], 0.f);
    }
  }
  __syncthreads();

  for (int idx = lane; idx < 48; idx += 32) {
    int m = idx / 3, o = idx - m * 3;
    float v = sB2[o];
#pragma unroll
    for (int k = 0; k < HID; ++k) v = fmaf((float)act[m * HID + k], sW2[k * 3 + o], v);
    if (active) out[(size_t)(tile * 16 + m) * 3 + o] = TWO_PI / (1.f + expf(-v));
  }
}

// ---------------------------------------------------------------------------
extern "C" void kernel_launch(void* const* d_in, const int* in_sizes, int n_in,
                              void* d_out, int out_size, void* d_ws, size_t ws_size,
                              hipStream_t stream) {
  const float*     node_f = (const float*)d_in[0];
  const long long* eidx   = (const long long*)d_in[1];
  const float*     edge_f = (const float*)d_in[2];
  const float* encW1 = (const float*)d_in[3];
  const float* encb1 = (const float*)d_in[4];
  const float* encW2 = (const float*)d_in[5];
  const float* encb2 = (const float*)d_in[6];
  const float* eencW = (const float*)d_in[7];
  const float* eencb = (const float*)d_in[8];
  const float* msgW1 = (const float*)d_in[9];
  const float* msgb1 = (const float*)d_in[10];
  const float* msgW2 = (const float*)d_in[11];
  const float* msgb2 = (const float*)d_in[12];
  const float* updW1 = (const float*)d_in[13];
  const float* updb1 = (const float*)d_in[14];
  const float* updW2 = (const float*)d_in[15];
  const float* updb2 = (const float*)d_in[16];
  const float* outW1 = (const float*)d_in[17];
  const float* outb1 = (const float*)d_in[18];
  const float* outW2 = (const float*)d_in[19];
  const float* outb2 = (const float*)d_in[20];

  char* ws = (char*)d_ws;
  auto carve = [&](size_t bytes) -> void* {
    void* p = (void*)ws;
    ws += (bytes + 255) & ~(size_t)255;
    return p;
  };
  _Float16* hA   = (_Float16*)carve((size_t)NN * HID * 2);
  _Float16* hB   = (_Float16*)carve((size_t)NN * HID * 2);
  _Float16* agg  = (_Float16*)carve((size_t)NN * HID * 2);   // f16 aggregation
  float*    invd = (float*)carve((size_t)NN * 4);
  _Float16* mW1f = (_Float16*)carve((size_t)3 * 8192 * 2);
  _Float16* mW2f = (_Float16*)carve((size_t)3 * 4096 * 2);
  _Float16* uW1f = (_Float16*)carve((size_t)3 * 8192 * 2);
  _Float16* uW2f = (_Float16*)carve((size_t)3 * 4096 * 2);
  _Float16* oW1f = (_Float16*)carve((size_t)4096 * 2);

  // weight fragment conversion
  for (int l = 0; l < 3; ++l) {
    convert_bfrag<<<32, 256, 0, stream>>>(msgW1 + (size_t)l * 128 * HID, mW1f + (size_t)l * 8192, 4);
    convert_bfrag<<<16, 256, 0, stream>>>(msgW2 + (size_t)l * HID * HID, mW2f + (size_t)l * 4096, 2);
    convert_bfrag<<<32, 256, 0, stream>>>(updW1 + (size_t)l * 128 * HID, uW1f + (size_t)l * 8192, 4);
    convert_bfrag<<<16, 256, 0, stream>>>(updW2 + (size_t)l * HID * HID, uW2f + (size_t)l * 4096, 2);
  }
  convert_bfrag<<<16, 256, 0, stream>>>(outW1, oW1f, 2);

  // encoder
  enc_kernel<<<(NN + 255) / 256, 256, 0, stream>>>(node_f, encW1, encb1, encW2, encb2, hA);

  // degree -> inverse denom
  zero_f32<<<(NN + 255) / 256, 256, 0, stream>>>(invd, NN);
  count_edges<<<(NE + 255) / 256, 256, 0, stream>>>(eidx + NE, invd);
  finalize_inv<<<(NN + 255) / 256, 256, 0, stream>>>(invd, NN);

  _Float16* hcur = hA;
  _Float16* hnxt = hB;
  const int nodeBlocks = (NN / 16 + 7) / 8;  // 391
  for (int l = 0; l < 3; ++l) {
    // zero f16 agg (N*HID halves = N*HID/2 dwords)
    zero_f32<<<(NN * HID / 2 + 255) / 256, 256, 0, stream>>>((float*)agg, NN * HID / 2);
    msg_kernel<<<NE / 128, 256, 0, stream>>>(
        hcur, eidx, eidx + NE, edge_f, eencW, eencb,
        mW1f + (size_t)l * 8192, msgb1 + (size_t)l * HID,
        mW2f + (size_t)l * 4096, msgb2 + (size_t)l * HID, agg);
    upd_kernel<<<nodeBlocks, 256, 0, stream>>>(
        hcur, agg, invd,
        uW1f + (size_t)l * 8192, updb1 + (size_t)l * HID,
        uW2f + (size_t)l * 4096, updb2 + (size_t)l * HID, hnxt);
    _Float16* t = hcur; hcur = hnxt; hnxt = t;
  }

  head_kernel<<<nodeBlocks, 256, 0, stream>>>(hcur, oW1f, outb1, outW2, outb2, (float*)d_out);
}
